// GraphSAGELayer_773094114149
// MI455X (gfx1250) — matrix operand order repaired and checked by reading, not compiled, not run on hardware
//
#include <hip/hip_runtime.h>
#include <cstddef>

// ---------------------------------------------------------------------------
// GraphSAGE layer for MI455X (gfx1250, wave32, WMMA).
//   agg  = (mask @ x) / max(deg,1)           [8192x8192 x 8192x512]
//   cat  = [x@W_self+b_self, agg@W_nb+b_nb]  [8192x1024]
//   out  = relu(cat @ W_comb + b_comb)       [8192x512]
// GEMMs run on v_wmma_f32_16x16x32_bf16 with bf16-split operands
// (mask exact in bf16; x/W use hi+lo splits). All LDS traffic is wide:
// fragments are ds_load_b128 pairs, tile fills are ds_store_b64.
// ---------------------------------------------------------------------------

typedef __bf16 bf16_t;
typedef __attribute__((ext_vector_type(16))) __bf16 v16bf;
typedef __attribute__((ext_vector_type(8)))  __bf16 v8bf;
typedef __attribute__((ext_vector_type(4)))  __bf16 v4bf;
typedef __attribute__((ext_vector_type(8)))  float  v8f;

#define GN    8192
#define GDIN  512
#define GDOUT 512
#define BM    128
#define BN    128
#define BK    32
#define APAD  8    // A-tile row stride = 40 bf16 = 80 B (16B-aligned)
#define BSLOT 24   // B-frag per-lane stride = 48 B (16B-aligned, bank-spread)

static __device__ inline bf16_t bf16_hi(float v) { return (bf16_t)v; }
static __device__ inline bf16_t bf16_lo(float v, bf16_t h) {
  return (bf16_t)(v - (float)h);
}

static __device__ inline v16bf cat8(v8bf a, v8bf b) {
  return __builtin_shufflevector(a, b, 0, 1, 2, 3, 4, 5, 6, 7,
                                 8, 9, 10, 11, 12, 13, 14, 15);
}

// A fragment: 16x32 bf16, rows m0..m0+15 of a row-major LDS tile.
// lane half (lane>=16) selects K run [half*16,+16) (matches B swizzle),
// contiguous in the row -> two ds_load_b128.
static __device__ inline v16bf load_a_frag(const bf16_t* lds, int stride,
                                           int m0, int lane) {
  const bf16_t* row = lds + (m0 + (lane & 15)) * stride + ((lane >> 4) << 4);
  const v8bf lo = *(const v8bf*)row;         // K = half*16 .. half*16+7
  const v8bf hi = *(const v8bf*)(row + 8);   // K = half*16+8 .. +7
  return cat8(lo, hi);
}

// B fragment from a pre-swizzled buffer: slot layout
//   addr = ((ntile*32 + (k>=16)*16 + (n&15)) * BSLOT + (k&15))
// so each lane's 16 values are contiguous -> two ds_load_b128.
static __device__ inline v16bf load_b_frag(const bf16_t* buf, int nt, int lane) {
  const bf16_t* p = buf + (nt * 32 + lane) * BSLOT;
  const v8bf lo = *(const v8bf*)p;        // k&15 = 0..7
  const v8bf hi = *(const v8bf*)(p + 8);  // k&15 = 8..15
  return cat8(lo, hi);
}

static __device__ inline v8f wmma_bf16(v16bf a, v16bf b, v8f c) {
  return __builtin_amdgcn_wmma_f32_16x16x32_bf16(
      /*neg_a=*/false, a, /*neg_b=*/false, b,
      /*c_mod=*/(short)0, c, /*reuse_a=*/false, /*reuse_b=*/false);
}

// Fill the swizzled B tile (BK x BN) from src (row-major, `ld` cols), rows
// k0..k0+31, cols n_blk..n_blk+127, splitting f32 -> bf16 hi/lo.
// Each thread: 16 coalesced b32 global loads, 8 ds_store_b64.
static __device__ inline void fill_b_tile(const float* __restrict__ src, int ld,
                                          int k0, int n_blk, int tid,
                                          bf16_t* __restrict__ sBh,
                                          bf16_t* __restrict__ sBl,
                                          int prefetch_next) {
  const int n  = tid & 127;
  const int kb = (tid >> 7) * 16;  // 0 or 16
  const int slotbase = ((n >> 4) * 32 + kb + (n & 15)) * BSLOT;
  const float* col = src + (size_t)(k0 + kb) * ld + n_blk + n;
#pragma unroll
  for (int g = 0; g < 4; ++g) {
    v4bf h, l;
#pragma unroll
    for (int kk = 0; kk < 4; ++kk) {
      const float f = col[(size_t)(g * 4 + kk) * ld];
      const bf16_t hh = bf16_hi(f);
      h[kk] = hh;
      l[kk] = bf16_lo(f, hh);
    }
    *(v4bf*)(sBh + slotbase + g * 4) = h;
    *(v4bf*)(sBl + slotbase + g * 4) = l;
  }
  if (prefetch_next)
    __builtin_prefetch(col + (size_t)BK * ld, 0, 1);  // global_prefetch_b8
}

// ---------------------------------------------------------------------------
// Kernel 1: per-row degree -> reciprocal of max(deg,1)
// ---------------------------------------------------------------------------
__global__ __launch_bounds__(256) void sage_deg_kernel(
    const int* __restrict__ adj, float* __restrict__ deg_inv) {
  __shared__ float red[256];
  const int row = blockIdx.x;
  const int4* arow = (const int4*)(adj + (size_t)row * GN);
  int s = 0;
  for (int j = threadIdx.x; j < GN / 4; j += 256) {
    const int4 v = arow[j];
    s += (v.x > 0) + (v.y > 0) + (v.z > 0) + (v.w > 0);
  }
  red[threadIdx.x] = (float)s;
  __syncthreads();
#pragma unroll
  for (int off = 128; off > 0; off >>= 1) {
    if (threadIdx.x < off) red[threadIdx.x] += red[threadIdx.x + off];
    __syncthreads();
  }
  if (threadIdx.x == 0) deg_inv[row] = 1.0f / fmaxf(red[0], 1.0f);
}

// ---------------------------------------------------------------------------
// Kernel 2: agg = (mask @ x) * deg_inv   (mask exact in bf16, x split hi/lo)
// grid x-major over column blocks so the 4 blocks sharing an adjacency row
// panel run adjacently -> adjacency hits L2, ~one HBM pass over 256 MB.
// ---------------------------------------------------------------------------
__global__ __launch_bounds__(256) void sage_agg_gemm(
    const int* __restrict__ adj, const float* __restrict__ x,
    const float* __restrict__ deg_inv, float* __restrict__ agg) {
  __shared__ bf16_t sM[BM][BK + APAD];            // adjacency mask tile
  __shared__ bf16_t sBh[(BN / 16) * 32 * BSLOT];  // x tile hi (swizzled)
  __shared__ bf16_t sBl[(BN / 16) * 32 * BSLOT];  // x tile lo (swizzled)

  const int tid   = threadIdx.x;
  const int lane  = tid & 31;
  const int wid   = tid >> 5;
  const int wm    = (wid & 3) * 32;   // 4 waves across rows
  const int wn    = (wid >> 2) * 64;  // 2 waves across cols
  const int m_blk = blockIdx.y * BM;
  const int n_blk = blockIdx.x * BN;

  v8f acc[2][4];
#pragma unroll
  for (int i = 0; i < 2; ++i)
#pragma unroll
    for (int j = 0; j < 4; ++j) acc[i][j] = (v8f)0.0f;

  for (int k0 = 0; k0 < GN; k0 += BK) {
    // --- adjacency tile: BM x BK int32 -> bf16 {0,1}, packed b64 stores ---
    {
      const int r = tid >> 3;       // 0..31
      const int c = (tid & 7) * 4;  // 0,4,..,28
#pragma unroll
      for (int p = 0; p < 4; ++p) {
        const int row = r + p * 32;
        const int4 v = *(const int4*)(adj + (size_t)(m_blk + row) * GN + k0 + c);
        v4bf m;
        m[0] = (bf16_t)(v.x > 0 ? 1.0f : 0.0f);
        m[1] = (bf16_t)(v.y > 0 ? 1.0f : 0.0f);
        m[2] = (bf16_t)(v.z > 0 ? 1.0f : 0.0f);
        m[3] = (bf16_t)(v.w > 0 ? 1.0f : 0.0f);
        *(v4bf*)&sM[row][c] = m;
      }
      if (k0 + BK < GN)
        __builtin_prefetch(adj + (size_t)(m_blk + r) * GN + k0 + BK + c, 0, 1);
    }
    // --- x tile -> swizzled hi/lo ---
    fill_b_tile(x, GDIN, k0, n_blk, tid, sBh, sBl, k0 + BK < GN);
    __syncthreads();

    v16bf am[2], bh[4], bl[4];
#pragma unroll
    for (int i = 0; i < 2; ++i)
      am[i] = load_a_frag(&sM[0][0], BK + APAD, wm + i * 16, lane);
#pragma unroll
    for (int j = 0; j < 4; ++j) {
      bh[j] = load_b_frag(sBh, (wn >> 4) + j, lane);
      bl[j] = load_b_frag(sBl, (wn >> 4) + j, lane);
    }
#pragma unroll
    for (int i = 0; i < 2; ++i)
#pragma unroll
      for (int j = 0; j < 4; ++j) {
        acc[i][j] = wmma_bf16(am[i], bh[j], acc[i][j]);
        acc[i][j] = wmma_bf16(am[i], bl[j], acc[i][j]);
      }
    __syncthreads();
  }

  // store: C/D layout — vgpr r, lane l -> m = (l>>4)*8 + r, n = l&15
#pragma unroll
  for (int i = 0; i < 2; ++i) {
#pragma unroll
    for (int r = 0; r < 8; ++r) {
      const int m = m_blk + wm + i * 16 + (lane >> 4) * 8 + r;
      const float dinv = deg_inv[m];
#pragma unroll
      for (int j = 0; j < 4; ++j) {
        const int n = n_blk + wn + j * 16 + (lane & 15);
        agg[(size_t)m * GDIN + n] = acc[i][j][r] * dinv;
      }
    }
  }
}

// ---------------------------------------------------------------------------
// Kernel 3 (generic): C = act(A[MxK] @ W[Kx512] + bias), 3-term bf16 split
// (Ahi*Whi + Alo*Whi + Ahi*Wlo) for ~1e-6 relative accuracy.
// ---------------------------------------------------------------------------
__global__ __launch_bounds__(256) void sage_feat_gemm(
    const float* __restrict__ A, int K, const float* __restrict__ W,
    const float* __restrict__ bias, float* __restrict__ C, int ldc, int relu) {
  __shared__ bf16_t sAh[BM][BK + APAD];
  __shared__ bf16_t sAl[BM][BK + APAD];
  __shared__ bf16_t sWh[(BN / 16) * 32 * BSLOT];
  __shared__ bf16_t sWl[(BN / 16) * 32 * BSLOT];

  const int tid   = threadIdx.x;
  const int lane  = tid & 31;
  const int wid   = tid >> 5;
  const int wm    = (wid & 3) * 32;
  const int wn    = (wid >> 2) * 64;
  const int m_blk = blockIdx.y * BM;
  const int n_blk = blockIdx.x * BN;

  v8f acc[2][4];
#pragma unroll
  for (int i = 0; i < 2; ++i)
#pragma unroll
    for (int j = 0; j < 4; ++j) acc[i][j] = (v8f)0.0f;

  for (int k0 = 0; k0 < K; k0 += BK) {
    // --- A tile: BM x BK f32 -> hi/lo, packed b64 stores ---
    {
      const int r = tid >> 3;
      const int c = (tid & 7) * 4;
#pragma unroll
      for (int p = 0; p < 4; ++p) {
        const int row = r + p * 32;
        const float4 v = *(const float4*)(A + (size_t)(m_blk + row) * K + k0 + c);
        v4bf h, l;
        bf16_t hh;
        hh = bf16_hi(v.x); h[0] = hh; l[0] = bf16_lo(v.x, hh);
        hh = bf16_hi(v.y); h[1] = hh; l[1] = bf16_lo(v.y, hh);
        hh = bf16_hi(v.z); h[2] = hh; l[2] = bf16_lo(v.z, hh);
        hh = bf16_hi(v.w); h[3] = hh; l[3] = bf16_lo(v.w, hh);
        *(v4bf*)&sAh[row][c] = h;
        *(v4bf*)&sAl[row][c] = l;
      }
      if (k0 + BK < K)
        __builtin_prefetch(A + (size_t)(m_blk + r) * K + k0 + BK + c, 0, 1);
    }
    // --- W tile -> swizzled hi/lo ---
    fill_b_tile(W, GDOUT, k0, n_blk, tid, sWh, sWl, k0 + BK < K);
    __syncthreads();

    v16bf ah[2], al[2], wh[4], wl[4];
#pragma unroll
    for (int i = 0; i < 2; ++i) {
      ah[i] = load_a_frag(&sAh[0][0], BK + APAD, wm + i * 16, lane);
      al[i] = load_a_frag(&sAl[0][0], BK + APAD, wm + i * 16, lane);
    }
#pragma unroll
    for (int j = 0; j < 4; ++j) {
      wh[j] = load_b_frag(sWh, (wn >> 4) + j, lane);
      wl[j] = load_b_frag(sWl, (wn >> 4) + j, lane);
    }
#pragma unroll
    for (int i = 0; i < 2; ++i)
#pragma unroll
      for (int j = 0; j < 4; ++j) {
        acc[i][j] = wmma_bf16(ah[i], wh[j], acc[i][j]);
        acc[i][j] = wmma_bf16(al[i], wh[j], acc[i][j]);
        acc[i][j] = wmma_bf16(ah[i], wl[j], acc[i][j]);
      }
    __syncthreads();
  }

#pragma unroll
  for (int i = 0; i < 2; ++i) {
#pragma unroll
    for (int j = 0; j < 4; ++j) {
      const int n = n_blk + wn + j * 16 + (lane & 15);
      const float b = bias[n];
#pragma unroll
      for (int r = 0; r < 8; ++r) {
        const int m = m_blk + wm + i * 16 + (lane >> 4) * 8 + r;
        float v = acc[i][j][r] + b;
        if (relu) v = fmaxf(v, 0.0f);
        C[(size_t)m * ldc + n] = v;
      }
    }
  }
}

// ---------------------------------------------------------------------------
extern "C" void kernel_launch(void* const* d_in, const int* in_sizes, int n_in,
                              void* d_out, int out_size, void* d_ws,
                              size_t ws_size, hipStream_t stream) {
  (void)in_sizes; (void)n_in; (void)out_size; (void)ws_size;
  const float* x      = (const float*)d_in[0];
  const int*   adj    = (const int*)d_in[1];
  const float* W_self = (const float*)d_in[2];
  const float* b_self = (const float*)d_in[3];
  const float* W_nb   = (const float*)d_in[4];
  const float* b_nb   = (const float*)d_in[5];
  const float* W_comb = (const float*)d_in[6];
  const float* b_comb = (const float*)d_in[7];
  float* out = (float*)d_out;

  // workspace: deg_inv (64KB aligned) | agg 16MB | cat 32MB  (~48MB total)
  char* ws = (char*)d_ws;
  float* deg_inv = (float*)ws;
  float* agg     = (float*)(ws + (64 << 10));
  float* cat     = (float*)(ws + (64 << 10) + ((size_t)16 << 20));

  sage_deg_kernel<<<GN, 256, 0, stream>>>(adj, deg_inv);

  sage_agg_gemm<<<dim3(GDIN / BN, GN / BM), 256, 0, stream>>>(adj, x, deg_inv,
                                                              agg);
  // cat[:, 0:512] = x @ W_self + b_self
  sage_feat_gemm<<<dim3(GDOUT / BN, GN / BM), 256, 0, stream>>>(
      x, GDIN, W_self, b_self, cat, 2 * GDOUT, 0);
  // cat[:, 512:1024] = agg @ W_nb + b_nb
  sage_feat_gemm<<<dim3(GDOUT / BN, GN / BM), 256, 0, stream>>>(
      agg, GDIN, W_nb, b_nb, cat + GDOUT, 2 * GDOUT, 0);
  // out = relu(cat @ W_comb + b_comb)
  sage_feat_gemm<<<dim3(GDOUT / BN, GN / BM), 256, 0, stream>>>(
      cat, 2 * GDOUT, W_comb, b_comb, out, GDOUT, 1);
}
